// RingModule_10505490006101
// MI455X (gfx1250) — matrix operand order repaired and checked by reading, not compiled, not run on hardware
//
#include <hip/hip_runtime.h>
#include <cmath>

typedef __attribute__((ext_vector_type(16))) __bf16 v16bf;
typedef __attribute__((ext_vector_type(8)))  float  v8f;

#define ALPHA 0.05f   // DT/TAU = 0.5/10
#define KCH   64      // K staged per LDS buffer (2 WMMAs per barrier)

// round-to-nearest-even f32 -> bf16 (upper 16 bits)
__device__ __forceinline__ unsigned short bf16_rne(float a) {
    unsigned u = __float_as_uint(a);
    u += 0x7FFFu + ((u >> 16) & 1u);
    return (unsigned short)(u >> 16);
}

// ---------------------------------------------------------------------------
// vel[t*B+b] = sum_i x[t,b,i] * w_in[i]
// ---------------------------------------------------------------------------
__global__ void vel_kernel(const float* __restrict__ x, const float* __restrict__ w_in,
                           float* __restrict__ vel, int TB, int IN) {
    int i = blockIdx.x * blockDim.x + threadIdx.x;
    if (i >= TB) return;
    const float* xr = x + (size_t)i * IN;
    float s = 0.f;
    for (int k = 0; k < IN; k += 4) {
        float4 xv = *(const float4*)(xr + k);
        float4 wv = *(const float4*)(w_in + k);
        s += xv.x * wv.x + xv.y * wv.y + xv.z * wv.z + xv.w * wv.w;
    }
    vel[i] = s;
}

// ---------------------------------------------------------------------------
// WT[j*H + k] = bf16(W[k*H + j])   (coalesced via LDS tile transpose)
// ---------------------------------------------------------------------------
__global__ void transpose_bf16_kernel(const float* __restrict__ W,
                                      unsigned short* __restrict__ WT, int H) {
    __shared__ unsigned short tile[32][33];
    int k0 = blockIdx.x * 32;
    int j0 = blockIdx.y * 32;
    for (int r = threadIdx.y; r < 32; r += 8)
        tile[r][threadIdx.x] = bf16_rne(W[(size_t)(k0 + r) * H + j0 + threadIdx.x]);
    __syncthreads();
    for (int r = threadIdx.y; r < 32; r += 8)
        WT[(size_t)(j0 + r) * H + k0 + threadIdx.x] = tile[threadIdx.x][r];
}

// ---------------------------------------------------------------------------
// hb[i] = bf16(h[i])   (one-time convert of hidden0)
// ---------------------------------------------------------------------------
__global__ void f32_to_bf16_kernel(const float* __restrict__ in,
                                   unsigned short* __restrict__ out, int n) {
    int i = blockIdx.x * blockDim.x + threadIdx.x;
    if (i < n) out[i] = bf16_rne(in[i]);
}

// ---------------------------------------------------------------------------
// WMMA helpers
// ---------------------------------------------------------------------------
__device__ __forceinline__ void loadB(const unsigned short* bp,
                                      uint4& b0, uint4& b1, uint4& b2, uint4& b3) {
    b0 = *(const uint4*)(bp);
    b1 = *(const uint4*)(bp + 16);
    b2 = *(const uint4*)(bp + 32);
    b3 = *(const uint4*)(bp + 48);
}

__device__ __forceinline__ void wmma_chunk(const unsigned int* arow, int half,
                                           const uint4& b0, const uint4& b1,
                                           const uint4& b2, const uint4& b3, v8f& c) {
    union { uint4 q[2]; v16bf v; } A0, A1, Bf0, Bf1;
    // sub-chunk s: A frag at dword s*16 + {4h, 8+4h}
    A0.q[0] = *(const uint4*)(arow + 4 * half);
    A0.q[1] = *(const uint4*)(arow + 8 + 4 * half);
    A1.q[0] = *(const uint4*)(arow + 16 + 4 * half);
    A1.q[1] = *(const uint4*)(arow + 24 + 4 * half);
    Bf0.q[0] = b0; Bf0.q[1] = b1;
    Bf1.q[0] = b2; Bf1.q[1] = b3;
    c = __builtin_amdgcn_wmma_f32_16x16x32_bf16(false, A0.v, false, Bf0.v, (short)0, c, false, false);
    c = __builtin_amdgcn_wmma_f32_16x16x32_bf16(false, A1.v, false, Bf1.v, (short)0, c, false, false);
}

// ---------------------------------------------------------------------------
// One recurrence step:
//   h_out = (1-ALPHA)*h_prev + ALPHA*relu(vel*gamma + 1 + h_prev @ W)
// 256 threads = 8 waves. blockIdx.y -> b-tile; blockIdx.x*8+wave -> h-tile.
// A comes from hb_in (bf16 copy of h_prev produced by previous step): staging
// is a pure b64 load + ds_store, no conversion VALU in the hot loop.
// Manual 2x unroll with alternating B-fragment register sets -> no reg copies,
// every WMMA consumes loads issued a full phase earlier.
// ---------------------------------------------------------------------------
__global__ __launch_bounds__(256)
void ring_step_kernel(const float* __restrict__ h_prev,          // [B,H] f32 (decay term)
                      const unsigned short* __restrict__ hb_in,  // [B,H] bf16 (matmul A)
                      const unsigned short* __restrict__ WT,     // [H,H] bf16 transposed
                      const float* __restrict__ vel_t,           // [B]
                      float* __restrict__ h_out,                 // [B,H] f32
                      unsigned short* __restrict__ hb_out,       // [B,H] bf16
                      int B, int H, int half_h)
{
    __shared__ __align__(16) unsigned int ldsA[2][16 * 36];  // 16 rows x 64 bf16, 144B stride

    const int tid  = threadIdx.x;
    const int wave = tid >> 5;
    const int lane = tid & 31;
    const int half = lane >> 4;
    const int ln   = lane & 15;

    const int bt    = blockIdx.y;
    const int ht    = blockIdx.x * 8 + wave;
    const int brow0 = bt * 16;

    // staging: thread i -> row = i>>4, bf16 cols [4c, 4c+4) : one b64 load+store
    const int srow = tid >> 4;
    const int scol = tid & 15;
    const unsigned short* __restrict__ aSrc =
        hb_in + (size_t)(brow0 + srow) * H + 4 * scol;
    const int sofs = srow * 36 + 2 * scol;     // dword offset within a buffer

    // B row pointer biased by the per-half K offset
    const unsigned short* __restrict__ bRow =
        WT + (size_t)(ht * 16 + ln) * H + 8 * half;

    const int NCH = H / KCH;   // assumed even, >= 4

    // ---- prologue: invariant at loop top (kk even):
    //   buf0 = chunk kk, aNext = A data chunk kk+1, aRegs = B chunk kk
    *(uint2*)&ldsA[0][sofs] = *(const uint2*)(aSrc);
    uint2 aNext = *(const uint2*)(aSrc + KCH);
    uint4 a0, a1, a2, a3, b0, b1, b2, b3;
    loadB(bRow, a0, a1, a2, a3);
    __syncthreads();

    v8f c = {};
    int kk = 0;
    for (; kk < NCH - 3; kk += 2) {
        // phase A: consume chunk kk (buf0, aRegs)
        *(uint2*)&ldsA[1][sofs] = aNext;                                 // stage kk+1
        aNext = *(const uint2*)(aSrc + (size_t)(kk + 2) * KCH);
        loadB(bRow + (size_t)(kk + 1) * KCH, b0, b1, b2, b3);
        wmma_chunk(&ldsA[0][ln * 36], half, a0, a1, a2, a3, c);
        __syncthreads();
        // phase B: consume chunk kk+1 (buf1, bRegs)
        *(uint2*)&ldsA[0][sofs] = aNext;                                 // stage kk+2
        aNext = *(const uint2*)(aSrc + (size_t)(kk + 3) * KCH);
        loadB(bRow + (size_t)(kk + 2) * KCH, a0, a1, a2, a3);
        wmma_chunk(&ldsA[1][ln * 36], half, b0, b1, b2, b3, c);
        __syncthreads();
    }
    // tail: kk == NCH-2 ; buf0 = chunk NCH-2, aNext = chunk NCH-1, aRegs = B chunk NCH-2
    *(uint2*)&ldsA[1][sofs] = aNext;                                     // stage NCH-1
    loadB(bRow + (size_t)(NCH - 1) * KCH, b0, b1, b2, b3);
    wmma_chunk(&ldsA[0][ln * 36], half, a0, a1, a2, a3, c);
    __syncthreads();
    wmma_chunk(&ldsA[1][ln * 36], half, b0, b1, b2, b3, c);

    // epilogue: slot v -> (M = v + 8*half, N = ln)
    const int j = ht * 16 + ln;
    const float g = (j < half_h) ? -1.0f : 1.0f;   // gamma, uniform per tile

    union { float4 f4[2]; float f[8]; } vv;
    const float* vp = vel_t + brow0 + 8 * half;
    vv.f4[0] = *(const float4*)(vp);
    vv.f4[1] = *(const float4*)(vp + 4);

    #pragma unroll
    for (int v = 0; v < 8; ++v) {
        const int b = brow0 + 8 * half + v;
        const size_t idx = (size_t)b * H + j;
        const float hp  = h_prev[idx];
        const float pre = vv.f[v] * g + 1.0f + c[v];
        const float act = pre > 0.f ? pre : 0.f;
        const float hn  = (1.0f - ALPHA) * hp + ALPHA * act;
        h_out[idx]  = hn;
        hb_out[idx] = bf16_rne(hn);
    }
}

// ---------------------------------------------------------------------------
extern "C" void kernel_launch(void* const* d_in, const int* in_sizes, int n_in,
                              void* d_out, int out_size, void* d_ws, size_t ws_size,
                              hipStream_t stream) {
    const float* x       = (const float*)d_in[0];   // [T,B,IN]
    const float* hidden0 = (const float*)d_in[1];   // [B,H]
    const float* W       = (const float*)d_in[2];   // [H,H]
    const float* w_in    = (const float*)d_in[3];   // [IN]

    const int IN = in_sizes[3];
    const int H  = (int)(sqrt((double)in_sizes[2]) + 0.5);
    const int B  = in_sizes[1] / H;
    const int T  = in_sizes[0] / (B * IN);
    const size_t BH = (size_t)B * H;

    float* out = (float*)d_out;
    // workspace layout
    char* ws = (char*)d_ws;
    unsigned short* WT  = (unsigned short*)ws;                 ws += (size_t)H * H * sizeof(unsigned short);
    float*          vel = (float*)ws;                          ws += (size_t)T * B * sizeof(float);
    unsigned short* hb0 = (unsigned short*)ws;                 ws += BH * sizeof(unsigned short);
    unsigned short* hb1 = (unsigned short*)ws;

    const int TB = T * B;
    vel_kernel<<<(TB + 255) / 256, 256, 0, stream>>>(x, w_in, vel, TB, IN);

    dim3 tb(32, 8), tg(H / 32, H / 32);
    transpose_bf16_kernel<<<tg, tb, 0, stream>>>(W, WT, H);

    f32_to_bf16_kernel<<<((int)BH + 255) / 256, 256, 0, stream>>>(hidden0, hb0, (int)BH);

    dim3 sb(256), sg(H / 128, B / 16);
    const float* hp = hidden0;
    unsigned short* hb[2] = { hb0, hb1 };
    for (int t = 0; t < T; ++t) {
        ring_step_kernel<<<sg, sb, 0, stream>>>(hp, hb[t & 1], WT, vel + (size_t)t * B,
                                                out + (size_t)t * BH, hb[(t + 1) & 1],
                                                B, H, H / 2);
        hp = out + (size_t)t * BH;
    }
    // hT = acts[T-1]
    hipMemcpyAsync(out + (size_t)T * BH, out + (size_t)(T - 1) * BH,
                   BH * sizeof(float), hipMemcpyDeviceToDevice, stream);
}
